// Translator_3496103379639
// MI455X (gfx1250) — compile-verified, matrix operand order checked by reading, not run on hardware
//
#include <hip/hip_runtime.h>
#include <hip/hip_bf16.h>
#include <stdint.h>

#define NSEG 16
#define HID  1024
#define MLPD 4096
#define NTOK 4096

#define BM 128
#define BN 128
#define BK 32
#define APITCH 40   // bf16 elems per LDS row: 32 data + 8 pad => 80B (16B-aligned, bank-staggered)

typedef __attribute__((ext_vector_type(16))) __bf16 bf16x16;
typedef __attribute__((ext_vector_type(8)))  __bf16 bf16x8;
typedef __attribute__((ext_vector_type(4)))  __bf16 bf16x4;
typedef __attribute__((ext_vector_type(8)))  float  f32x8;
typedef __attribute__((ext_vector_type(4)))  float  f32x4;

// ---------- fp32 -> bf16 hi/lo split via packed hardware converts (RNE) ----------
__device__ __forceinline__ void split_store4(__bf16* __restrict__ sh,
                                             __bf16* __restrict__ sl,
                                             int o, f32x4 f) {
  bf16x4 h = __builtin_convertvector(f, bf16x4);              // v_cvt_pk_bf16_f32
  f32x4  r = f - __builtin_convertvector(h, f32x4);           // exact residual
  bf16x4 l = __builtin_convertvector(r, bf16x4);
  *(bf16x4*)(sh + o) = h;   // 8B LDS stores: row*80 + {0,8,...,56}
  *(bf16x4*)(sl + o) = l;
}

// Load one 16x32 bf16 fragment in the documented per-lane layout:
// lanes 0-15: M=lane, K = 0..7 then 16..23 ; lanes 16-31: M=lane-16, K = 8..15 then 24..31
__device__ __forceinline__ bf16x16 load_frag(const __bf16* __restrict__ tile,
                                             int rowOff, int lane) {
  int r  = rowOff + (lane & 15);
  int kb = (lane >> 4) << 3;                  // 0 or 8
  const __bf16* p = tile + r * APITCH + kb;
  bf16x8 lo = *(const bf16x8*)(p);            // K = kb .. kb+7
  bf16x8 hi = *(const bf16x8*)(p + 16);       // K = kb+16 .. kb+23
  return __builtin_shufflevector(lo, hi, 0,1,2,3,4,5,6,7,8,9,10,11,12,13,14,15);
}

// ---------- token counting sort ----------
__global__ void k_zero(int* c) { if (threadIdx.x < 48) c[threadIdx.x] = 0; }

__global__ void k_count(const int* __restrict__ seg, int* __restrict__ cnt, int n) {
  int i = blockIdx.x * blockDim.x + threadIdx.x;
  if (i < n) atomicAdd(&cnt[seg[i]], 1);
}

__global__ void k_scan(const int* __restrict__ cnt, int* __restrict__ start, int* __restrict__ fill) {
  if (threadIdx.x == 0) {
    int acc = 0;
    for (int s = 0; s < NSEG; ++s) { start[s] = acc; fill[s] = acc; acc += cnt[s]; }
  }
}

__global__ void k_scatter(const int* __restrict__ seg, int* __restrict__ fill,
                          int* __restrict__ perm, int n) {
  int i = blockIdx.x * blockDim.x + threadIdx.x;
  if (i < n) { int p = atomicAdd(&fill[seg[i]], 1); perm[p] = i; }
}

// ---------- per-expert GEMM ----------
// MODE 0: inter[ord][n] = relu( X[perm[ord]] . W1[s][n] + b1[s][n] )
// MODE 1: out[perm[ord]][n] = inter[ord] . W2[s][n] + b2[s][n]
template<int MODE>
__global__ __launch_bounds__(256)
void expert_gemm(const float* __restrict__ A,     // [NTOK][K] (X or inter)
                 const float* __restrict__ W,     // [NSEG][N][K]
                 const float* __restrict__ bias,  // [NSEG][N]
                 float* __restrict__ Out,         // [NTOK][N]
                 const int* __restrict__ perm,
                 const int* __restrict__ segStart,
                 const int* __restrict__ segCnt,
                 int K, int N, int NBN, int MSLOTS)
{
  __shared__ __align__(16) __bf16 sAh[BM * APITCH];
  __shared__ __align__(16) __bf16 sAl[BM * APITCH];
  __shared__ __align__(16) __bf16 sBh[BN * APITCH];
  __shared__ __align__(16) __bf16 sBl[BN * APITCH];

  const int s     = blockIdx.y;
  const int nb    = blockIdx.x % NBN;
  const int mslot = blockIdx.x / NBN;
  const int cnt   = segCnt[s];
  const int base  = segStart[s];
  const int tid   = threadIdx.x;
  const int lane  = tid & 31;
  const int wave  = tid >> 5;
  const int wm    = (wave & 1) * 64;   // wave's M offset (64 rows)
  const int wn    = (wave >> 1) * 32;  // wave's N offset (32 cols)
  const int nTile = nb * BN;
  const int kTiles = K / BK;
  const int srow  = tid >> 3;          // staging row within 32-row pass
  const int scol  = (tid & 7) * 4;     // staging column (4 floats)

  for (int mt = mslot; mt * BM < cnt; mt += MSLOTS) {
    // ---- hoisted per-thread source pointers (perm gather done ONCE per m-tile) ----
    const float* aptr[4];
    bool         aval[4];
    const float* wptr[4];
    #pragma unroll
    for (int rr = 0; rr < 4; ++rr) {
      int row = rr * 32 + srow;
      int gm  = mt * BM + row;
      aval[rr] = (gm < cnt);
      long arow = 0;
      if (aval[rr]) arow = (MODE == 0) ? (long)perm[base + gm] : (long)(base + gm);
      aptr[rr] = A + arow * (long)K + scol;
      wptr[rr] = W + ((long)s * N + (long)(nTile + row)) * (long)K + scol;
    }

    // ---- software pipeline: global tile (kt) lives in registers while kt-1 computes ----
    f32x4 va[4], vb[4];
    auto fetch = [&](int kt) {
      const f32x4 z = {0.f, 0.f, 0.f, 0.f};
      #pragma unroll
      for (int rr = 0; rr < 4; ++rr) {
        va[rr] = aval[rr] ? *(const f32x4*)(aptr[rr] + (long)kt * BK) : z;
        vb[rr] = *(const f32x4*)(wptr[rr] + (long)kt * BK);
      }
    };
    fetch(0);

    f32x8 acc[4][2];
    #pragma unroll
    for (int i = 0; i < 4; ++i)
      #pragma unroll
      for (int j = 0; j < 2; ++j)
        acc[i][j] = (f32x8){0.f,0.f,0.f,0.f,0.f,0.f,0.f,0.f};

    for (int kt = 0; kt < kTiles; ++kt) {
      __syncthreads();   // previous iteration's readers done
      #pragma unroll
      for (int rr = 0; rr < 4; ++rr) {
        int o = (rr * 32 + srow) * APITCH + scol;
        split_store4(sAh, sAl, o, va[rr]);
        split_store4(sBh, sBl, o, vb[rr]);
      }
      __syncthreads();
      if (kt + 1 < kTiles) fetch(kt + 1);   // overlap next global tile with WMMAs

      // Fragments + 3-product bf16 split WMMA (hi*hi + hi*lo + lo*hi ~= fp32)
      bf16x16 bh[2], bl[2];
      #pragma unroll
      for (int nf = 0; nf < 2; ++nf) {
        bh[nf] = load_frag(sBh, wn + nf * 16, lane);
        bl[nf] = load_frag(sBl, wn + nf * 16, lane);
      }
      #pragma unroll
      for (int mf = 0; mf < 4; ++mf) {
        bf16x16 ah = load_frag(sAh, wm + mf * 16, lane);
        bf16x16 al = load_frag(sAl, wm + mf * 16, lane);
        #pragma unroll
        for (int nf = 0; nf < 2; ++nf) {
          acc[mf][nf] = __builtin_amdgcn_wmma_f32_16x16x32_bf16(
              false, al, false, bh[nf], (short)0, acc[mf][nf], false, false);
          acc[mf][nf] = __builtin_amdgcn_wmma_f32_16x16x32_bf16(
              false, ah, false, bl[nf], (short)0, acc[mf][nf], false, false);
          acc[mf][nf] = __builtin_amdgcn_wmma_f32_16x16x32_bf16(
              false, ah, false, bh[nf], (short)0, acc[mf][nf], false, false);
        }
      }
    }

    // Epilogue: bias (+relu for MODE 0), masked store.
    // C/D layout: VGPR e -> M = e + 8*(lane>=16), N = lane&15.
    #pragma unroll
    for (int mf = 0; mf < 4; ++mf) {
      #pragma unroll
      for (int nf = 0; nf < 2; ++nf) {
        int n = nTile + wn + nf * 16 + (lane & 15);
        float bv = bias[(long)s * N + n];
        #pragma unroll
        for (int e = 0; e < 8; ++e) {
          int m  = wm + mf * 16 + e + ((lane >> 4) << 3);
          int gm = mt * BM + m;
          if (gm < cnt) {
            float v = acc[mf][nf][e] + bv;
            if (MODE == 0) v = fmaxf(v, 0.f);
            long orow = (MODE == 0) ? (long)(base + gm) : (long)perm[base + gm];
            Out[orow * (long)N + n] = v;
          }
        }
      }
    }
  }
}

extern "C" void kernel_launch(void* const* d_in, const int* in_sizes, int n_in,
                              void* d_out, int out_size, void* d_ws, size_t ws_size,
                              hipStream_t stream) {
  (void)in_sizes; (void)n_in; (void)out_size; (void)ws_size;
  const float* X  = (const float*)d_in[0];   // [4096][1024]
  const int*   SG = (const int*)d_in[1];     // [4096]
  const float* W1 = (const float*)d_in[2];   // [16][4096][1024]
  const float* W2 = (const float*)d_in[3];   // [16][1024][4096]
  const float* B1 = (const float*)d_in[4];   // [16][4096]
  const float* B2 = (const float*)d_in[5];   // [16][1024]
  float* OUT = (float*)d_out;                // [4096][1024]

  // Workspace layout: counters/perm in first 64KB, fp32 inter after.
  int*   cnt   = (int*)d_ws;        // 16
  int*   start = cnt + 16;          // 16
  int*   fill  = cnt + 32;          // 16
  int*   perm  = cnt + 64;          // 4096
  float* inter = (float*)((char*)d_ws + (1 << 16));  // [4096][4096] = 64 MB

  // Counting sort of tokens by segment (output is perm-order invariant).
  k_zero   <<<1, 64, 0, stream>>>(cnt);
  k_count  <<<NTOK / 256, 256, 0, stream>>>(SG, cnt, NTOK);
  k_scan   <<<1, 32, 0, stream>>>(cnt, start, fill);
  k_scatter<<<NTOK / 256, 256, 0, stream>>>(SG, fill, perm, NTOK);

  // GEMM1: inter = relu(Xg @ W1^T + b1)   (N=4096, K=1024)
  expert_gemm<0><<<dim3((MLPD / BN) * 2, NSEG), 256, 0, stream>>>(
      X, W1, B1, inter, perm, start, cnt, HID, MLPD, MLPD / BN, 2);
  // GEMM2: out[perm] = inter @ W2^T + b2  (N=1024, K=4096)
  expert_gemm<1><<<dim3((HID / BN) * 4, NSEG), 256, 0, stream>>>(
      inter, W2, B2, OUT, perm, start, cnt, MLPD, HID, HID / BN, 4);
}
